// FeatureFusionSegNetwork_68882685493938
// MI455X (gfx1250) — compile-verified, matrix operand order checked by reading, not run on hardware
//
#include <hip/hip_runtime.h>
#include <hip/hip_bf16.h>

// ---------------------------------------------------------------------------
// CDNA5 (gfx1250) FeatureFusionSegNetwork forward.
// All heavy math through v_wmma_f32_16x16x32_f16 (wave32 WMMA), 4 accumulator
// tiles per wave for operand reuse, zero-page pointer-select for SAME padding.
// Latency hiding via wave-level parallelism (plain k-loop; manual ping-pong
// pipelining was tried and caused accumulator copies + hazard nops).
// ---------------------------------------------------------------------------

typedef __attribute__((ext_vector_type(16))) _Float16 v16h;
typedef __attribute__((ext_vector_type(8)))  _Float16 v8h;
typedef __attribute__((ext_vector_type(4)))  _Float16 v4h;
typedef __attribute__((ext_vector_type(8)))  float    v8f;

union V16H { v16h v; v8h h[2]; };

#define B_   16
#define CIN  1024
#define CK   256
#define CV   512
#define HW   1024   // 32x32
#define WDIM 32

#define WMMA_F16(A, Bf, C) \
    __builtin_amdgcn_wmma_f32_16x16x32_f16(false, (A), false, (Bf), (short)0, (C), false, false)

// Per-lane fragment load from row-major [rows][Ktotal] f16; rowbase = lane row.
// Element e -> K = (e<8) ? g*8+e : 16+g*8+(e-8)   (ISA 7.12.2 16-bit layout)
__device__ __forceinline__ v16h frag_k32(const _Float16* __restrict__ rowbase,
                                         int k0, int g) {
    V16H f;
    f.h[0] = *(const v8h*)(rowbase + k0 + g * 8);
    f.h[1] = *(const v8h*)(rowbase + k0 + 16 + g * 8);
    return f.v;
}

// ---------------------------------------------------------------------------
// 0) Zero the padding page (2 KB) used by conv boundary lanes.
// ---------------------------------------------------------------------------
__global__ __launch_bounds__(256) void zero_fill(_Float16* __restrict__ p) {
    p[blockIdx.x * 256 + threadIdx.x] = (_Float16)0.0f;
}

// ---------------------------------------------------------------------------
// 1) NCHW f32 -> NHWC f16 transpose/convert (LDS-tiled, coalesced both ways)
//    grid (32, 32, B), block (32, 8)
// ---------------------------------------------------------------------------
__global__ __launch_bounds__(256) void nchw_to_nhwc_f16(
    const float* __restrict__ x, _Float16* __restrict__ o) {
    __shared__ float t[32][33];
    const int p0 = blockIdx.x * 32;
    const int c0 = blockIdx.y * 32;
    const int b  = blockIdx.z;
    const int tx = threadIdx.x, ty = threadIdx.y;
#pragma unroll
    for (int i = 0; i < 4; ++i) {
        int c = ty * 4 + i;
        t[c][tx] = x[((size_t)(b * CIN + c0 + c)) * HW + p0 + tx];
    }
    __syncthreads();
#pragma unroll
    for (int i = 0; i < 4; ++i) {
        int p = ty * 4 + i;
        o[((size_t)(b * HW + p0 + p)) * CIN + c0 + tx] = (_Float16)t[tx][p];
    }
}

// ---------------------------------------------------------------------------
// 2) Prepack conv weights [Cout][Cin][3][3] f32 into per-lane WMMA A-fragment
//    order pk[ct][tap][kc][lane][e] (f16). One thread per packed element.
// ---------------------------------------------------------------------------
__global__ __launch_bounds__(256) void pack_weights(
    const float* __restrict__ w, _Float16* __restrict__ pk, int total) {
    int idx = blockIdx.x * 256 + threadIdx.x;
    if (idx >= total) return;
    int e    = idx & 15;
    int lane = (idx >> 4) & 31;
    int kc   = (idx >> 9) & 31;
    int rest = idx >> 14;              // 16*32*32 = 16384
    int tap  = rest % 9;
    int ct   = rest / 9;
    int m = lane & 15, g = lane >> 4;
    int k = (e < 8) ? (g * 8 + e) : (16 + g * 8 + (e - 8));
    int cout = ct * 16 + m;
    int ci   = kc * 32 + k;
    pk[idx] = (_Float16)w[((size_t)cout * CIN + ci) * 9 + tap];
}

// ---------------------------------------------------------------------------
// 3) Implicit-GEMM 3x3 SAME conv. Each wave: 64 Cout x 16 pixels (4 acc
//    tiles), B fragment loaded once per k-step and reused by 4 WMMAs.
//    Padding via zero-page pointer select (no per-iteration cndmask).
//    grid (64, Cout/256, B), block 128.
//    mode 0: f16 [B][HW][Cout] | mode 1: f16 [B][Cout][HW] | mode 2: f32 d_out
// ---------------------------------------------------------------------------
__global__ __launch_bounds__(128) void conv3x3_wmma(
    const _Float16* __restrict__ x,    // [B][HW][CIN] f16
    const _Float16* __restrict__ wpk,  // packed weights
    const _Float16* __restrict__ zpage,
    const float*    __restrict__ bias,
    int Cout, int mode,
    _Float16* __restrict__ out_h, float* __restrict__ out_f) {
    const int lane = threadIdx.x & 31;
    const int wave = threadIdx.x >> 5;
    const int n = lane & 15;
    const int g = lane >> 4;
    const int p0 = blockIdx.x * 16;
    const int y  = p0 >> 5;
    const int x0 = p0 & 31;
    const int ct0 = (blockIdx.y * 4 + wave) * 4;   // 4 consecutive cout tiles
    const int b = blockIdx.z;

    v8f acc0 = {}, acc1 = {}, acc2 = {}, acc3 = {};
    const size_t wstride = (size_t)9 * 32 * 512;   // halves per cout tile
#pragma unroll
    for (int tap = 0; tap < 9; ++tap) {
        const int dy = tap / 3 - 1, dx = tap % 3 - 1;
        const int sy = y + dy;
        const int sx = x0 + n + dx;
        const bool valid = (sy >= 0) & (sy < WDIM) & (sx >= 0) & (sx < WDIM);
        // pointer select once per tap: invalid lanes stream zeros
        const _Float16* xb =
            valid ? (x + ((size_t)(b * HW + sy * WDIM + sx)) * CIN) : zpage;
        const _Float16* wt0 = wpk + (size_t)(ct0 + 0) * wstride + (size_t)tap * 32 * 512;
        const _Float16* wt1 = wt0 + wstride;
        const _Float16* wt2 = wt1 + wstride;
        const _Float16* wt3 = wt2 + wstride;
        for (int kc = 0; kc < 32; ++kc) {
            const int c0 = kc * 32;
            V16H bf;
            bf.h[0] = *(const v8h*)(xb + c0 + g * 8);
            bf.h[1] = *(const v8h*)(xb + c0 + 16 + g * 8);
            const int wo = kc * 512 + lane * 16;
            V16H a0, a1, a2, a3;
            a0.h[0] = *(const v8h*)(wt0 + wo); a0.h[1] = *(const v8h*)(wt0 + wo + 8);
            a1.h[0] = *(const v8h*)(wt1 + wo); a1.h[1] = *(const v8h*)(wt1 + wo + 8);
            a2.h[0] = *(const v8h*)(wt2 + wo); a2.h[1] = *(const v8h*)(wt2 + wo + 8);
            a3.h[0] = *(const v8h*)(wt3 + wo); a3.h[1] = *(const v8h*)(wt3 + wo + 8);
            acc0 = WMMA_F16(a0.v, bf.v, acc0);
            acc1 = WMMA_F16(a1.v, bf.v, acc1);
            acc2 = WMMA_F16(a2.v, bf.v, acc2);
            acc3 = WMMA_F16(a3.v, bf.v, acc3);
        }
    }

    const int pix = p0 + n;
    v8f accs[4] = {acc0, acc1, acc2, acc3};
#pragma unroll
    for (int t = 0; t < 4; ++t) {
        const int cout0 = (ct0 + t) * 16;
        if (mode == 0) {               // [B][HW][Cout]: 8 contiguous halves
            v8h o;
#pragma unroll
            for (int r = 0; r < 8; ++r)
                o[r] = (_Float16)(accs[t][r] + bias[cout0 + 8 * g + r]);
            *(v8h*)(out_h + ((size_t)(b * HW + pix)) * Cout + cout0 + 8 * g) = o;
        } else if (mode == 1) {        // [B][Cout][HW]
#pragma unroll
            for (int r = 0; r < 8; ++r) {
                int cout = cout0 + 8 * g + r;
                out_h[((size_t)(b * Cout + cout)) * HW + pix] =
                    (_Float16)(accs[t][r] + bias[cout]);
            }
        } else {                       // f32 into d_out concat slot (+512 ch)
#pragma unroll
            for (int r = 0; r < 8; ++r) {
                int cout = cout0 + 8 * g + r;
                out_f[((size_t)(b * 1024 + 512 + cout)) * HW + pix] =
                    accs[t][r] + bias[cout];
            }
        }
    }
}

// ---------------------------------------------------------------------------
// 4) Score GEMM: S_t[b][q][m] = (1/16) * sum_c qk_t[b][q][c] * mk_t[b][m][c]
//    K=256. Wave: 16 q x 64 m (A fragment reused by 4 WMMAs).
//    grid (4, 64, B), block 128.
// ---------------------------------------------------------------------------
__global__ __launch_bounds__(128) void score_gemm(
    const _Float16* __restrict__ qk, const _Float16* __restrict__ mk,
    float* __restrict__ sc) {
    const int lane = threadIdx.x & 31;
    const int wave = threadIdx.x >> 5;
    const int n = lane & 15, g = lane >> 4;
    const int mbase = (blockIdx.x * 4 + wave) * 64;
    const int q0 = blockIdx.y * 16;
    const int b = blockIdx.z;
    const _Float16* arow = qk + ((size_t)(b * HW + q0 + n)) * CK;
    const _Float16* br0 = mk + ((size_t)(b * HW + mbase + n)) * CK;
    const _Float16* br1 = br0 + (size_t)16 * CK;
    const _Float16* br2 = br1 + (size_t)16 * CK;
    const _Float16* br3 = br2 + (size_t)16 * CK;
    v8f acc0 = {}, acc1 = {}, acc2 = {}, acc3 = {};
#pragma unroll
    for (int kc = 0; kc < 8; ++kc) {
        v16h a = frag_k32(arow, kc * 32, g);
        acc0 = WMMA_F16(a, frag_k32(br0, kc * 32, g), acc0);
        acc1 = WMMA_F16(a, frag_k32(br1, kc * 32, g), acc1);
        acc2 = WMMA_F16(a, frag_k32(br2, kc * 32, g), acc2);
        acc3 = WMMA_F16(a, frag_k32(br3, kc * 32, g), acc3);
    }
    v8f accs[4] = {acc0, acc1, acc2, acc3};
#pragma unroll
    for (int t = 0; t < 4; ++t)
#pragma unroll
        for (int r = 0; r < 8; ++r)
            sc[((size_t)(b * HW + q0 + r + 8 * g)) * HW + mbase + t * 16 + n] =
                accs[t][r] * 0.0625f;
}

// ---------------------------------------------------------------------------
// 5) Softmax over m (contiguous rows of 1024 f32) -> normalized f16.
//    grid B*HW, block 256; LDS tree reductions.
// ---------------------------------------------------------------------------
__global__ __launch_bounds__(256) void softmax_rows(
    const float* __restrict__ sc, _Float16* __restrict__ wsm) {
    const int row = blockIdx.x, tid = threadIdx.x;
    const float4* sp = (const float4*)(sc + (size_t)row * HW);
    float4 v = sp[tid];
    __shared__ float red[256];
    red[tid] = fmaxf(fmaxf(v.x, v.y), fmaxf(v.z, v.w));
    __syncthreads();
    for (int s = 128; s > 0; s >>= 1) {
        if (tid < s) red[tid] = fmaxf(red[tid], red[tid + s]);
        __syncthreads();
    }
    const float m = red[0];
    __syncthreads();
    float e0 = __expf(v.x - m), e1 = __expf(v.y - m);
    float e2 = __expf(v.z - m), e3 = __expf(v.w - m);
    red[tid] = e0 + e1 + e2 + e3;
    __syncthreads();
    for (int s = 128; s > 0; s >>= 1) {
        if (tid < s) red[tid] += red[tid + s];
        __syncthreads();
    }
    const float inv = 1.0f / red[0];
    v4h o;
    o[0] = (_Float16)(e0 * inv); o[1] = (_Float16)(e1 * inv);
    o[2] = (_Float16)(e2 * inv); o[3] = (_Float16)(e3 * inv);
    *(v4h*)(wsm + (size_t)row * HW + tid * 4) = o;
}

// ---------------------------------------------------------------------------
// 6) Attention GEMM: out[b][cv][q] = sum_m mv[b][cv][m] * wsm[b][q][m]
//    K=1024. Wave: 64 cv x 16 q (B fragment reused by 4 WMMAs).
//    grid (2, 64, B), block 128; f32 straight into d_out.
// ---------------------------------------------------------------------------
__global__ __launch_bounds__(128) void attn_gemm(
    const _Float16* __restrict__ mv, const _Float16* __restrict__ wsm,
    float* __restrict__ out) {
    const int lane = threadIdx.x & 31;
    const int wave = threadIdx.x >> 5;
    const int n = lane & 15, g = lane >> 4;
    const int cvbase = (blockIdx.x * 4 + wave) * 64;
    const int q0 = blockIdx.y * 16;
    const int b = blockIdx.z;
    const _Float16* ar0 = mv + ((size_t)(b * CV + cvbase + n)) * HW;
    const _Float16* ar1 = ar0 + (size_t)16 * HW;
    const _Float16* ar2 = ar1 + (size_t)16 * HW;
    const _Float16* ar3 = ar2 + (size_t)16 * HW;
    const _Float16* brow = wsm + ((size_t)(b * HW + q0 + n)) * HW;
    v8f acc0 = {}, acc1 = {}, acc2 = {}, acc3 = {};
    for (int kc = 0; kc < 32; ++kc) {
        v16h bf = frag_k32(brow, kc * 32, g);
        acc0 = WMMA_F16(frag_k32(ar0, kc * 32, g), bf, acc0);
        acc1 = WMMA_F16(frag_k32(ar1, kc * 32, g), bf, acc1);
        acc2 = WMMA_F16(frag_k32(ar2, kc * 32, g), bf, acc2);
        acc3 = WMMA_F16(frag_k32(ar3, kc * 32, g), bf, acc3);
    }
    v8f accs[4] = {acc0, acc1, acc2, acc3};
#pragma unroll
    for (int t = 0; t < 4; ++t)
#pragma unroll
        for (int r = 0; r < 8; ++r)
            out[((size_t)(b * 1024 + cvbase + t * 16 + r + 8 * g)) * HW + q0 + n] =
                accs[t][r];
}

// ---------------------------------------------------------------------------
// Host-side launch sequence (graph-capture safe: plain stream launches only)
// ---------------------------------------------------------------------------
extern "C" void kernel_launch(void* const* d_in, const int* in_sizes, int n_in,
                              void* d_out, int out_size, void* d_ws, size_t ws_size,
                              hipStream_t stream) {
    const float* src_temp   = (const float*)d_in[0];
    const float* src_search = (const float*)d_in[1];
    const float* wk_m = (const float*)d_in[2];
    const float* bk_m = (const float*)d_in[3];
    const float* wv_m = (const float*)d_in[4];
    const float* bv_m = (const float*)d_in[5];
    const float* wk_q = (const float*)d_in[6];
    const float* bk_q = (const float*)d_in[7];
    const float* wv_q = (const float*)d_in[8];
    const float* bv_q = (const float*)d_in[9];
    float* out = (float*)d_out;

    char* ws = (char*)d_ws;
    size_t off = 0;
    _Float16* xT   = (_Float16*)(ws + off); off += (size_t)B_ * HW * CIN * 2;
    _Float16* xS   = (_Float16*)(ws + off); off += (size_t)B_ * HW * CIN * 2;
    _Float16* pkm  = (_Float16*)(ws + off); off += (size_t)CK * CIN * 9 * 2;
    _Float16* pvm  = (_Float16*)(ws + off); off += (size_t)CV * CIN * 9 * 2;
    _Float16* pkq  = (_Float16*)(ws + off); off += (size_t)CK * CIN * 9 * 2;
    _Float16* pvq  = (_Float16*)(ws + off); off += (size_t)CV * CIN * 9 * 2;
    _Float16* mk_t = (_Float16*)(ws + off); off += (size_t)B_ * HW * CK * 2;
    _Float16* qk_t = (_Float16*)(ws + off); off += (size_t)B_ * HW * CK * 2;
    _Float16* mv   = (_Float16*)(ws + off); off += (size_t)B_ * CV * HW * 2;
    float*    sc   = (float*)   (ws + off); off += (size_t)B_ * HW * HW * 4;
    _Float16* wsm  = (_Float16*)(ws + off); off += (size_t)B_ * HW * HW * 2;
    _Float16* zpg  = (_Float16*)(ws + off); off += (size_t)CIN * 2;  // zero page

    // 0) zero pad page, 1) layout conversion
    zero_fill<<<CIN / 256, 256, 0, stream>>>(zpg);
    dim3 tg(32, 32, B_), tb(32, 8);
    nchw_to_nhwc_f16<<<tg, tb, 0, stream>>>(src_temp, xT);
    nchw_to_nhwc_f16<<<tg, tb, 0, stream>>>(src_search, xS);

    // 2) weight prepack
    int tk = CK * CIN * 9, tv = CV * CIN * 9;
    pack_weights<<<tk / 256, 256, 0, stream>>>(wk_m, pkm, tk);
    pack_weights<<<tv / 256, 256, 0, stream>>>(wv_m, pvm, tv);
    pack_weights<<<tk / 256, 256, 0, stream>>>(wk_q, pkq, tk);
    pack_weights<<<tv / 256, 256, 0, stream>>>(wv_q, pvq, tv);

    // 3) four convs (each wave: 64 cout x 16 pixels)
    conv3x3_wmma<<<dim3(64, CK / 256, B_), 128, 0, stream>>>(xT, pkm, zpg, bk_m, CK, 0, mk_t, nullptr);
    conv3x3_wmma<<<dim3(64, CV / 256, B_), 128, 0, stream>>>(xT, pvm, zpg, bv_m, CV, 1, mv,   nullptr);
    conv3x3_wmma<<<dim3(64, CK / 256, B_), 128, 0, stream>>>(xS, pkq, zpg, bk_q, CK, 0, qk_t, nullptr);
    conv3x3_wmma<<<dim3(64, CV / 256, B_), 128, 0, stream>>>(xS, pvq, zpg, bv_q, CV, 2, nullptr, out);

    // 4) scores (scaled), 5) softmax over m, 6) attention GEMM -> mem_info
    score_gemm<<<dim3(4, 64, B_), 128, 0, stream>>>(qk_t, mk_t, sc);
    softmax_rows<<<B_ * HW, 256, 0, stream>>>(sc, wsm);
    attn_gemm<<<dim3(2, 64, B_), 128, 0, stream>>>(mv, wsm, out);
}
